// NeuralFieldModel_45801531244981
// MI455X (gfx1250) — matrix operand order repaired and checked by reading, not compiled, not run on hardware
//
#include <hip/hip_runtime.h>
#include <hip/hip_bf16.h>

typedef __attribute__((ext_vector_type(16))) _Float16 v16h;
typedef __attribute__((ext_vector_type(8)))  _Float16 h8;
typedef __attribute__((ext_vector_type(8)))  float    v8f;

#define FDIM      32
#define GRIDN     129
#define HID       128
#define DIN       35          // 3 + FDIM
#define KPAD      64          // K padded for 2x wmma 16x16x32
#define EXSTRIDE  72          // LDS row stride in halves (144B, 16B aligned, bank-friendly)
#define DINV      (1.0f/1.2f) // 1/DOMAIN

static constexpr long VOXELS = (long)GRIDN * GRIDN * GRIDN;

// ---------------------------------------------------------------------------
// Pass 1: transpose fm [32][129^3] f32  ->  ws [129^3][32] f16 (channel-last)
// Turns the per-point gather from 256 scattered 4B loads into 8x64B loads.
// ---------------------------------------------------------------------------
__global__ void __launch_bounds__(256)
fm_transpose_f16(const float* __restrict__ fm, _Float16* __restrict__ ws) {
    long v = (long)blockIdx.x * 256 + threadIdx.x;
    if (v >= VOXELS) return;
    h8 o[4];
#pragma unroll
    for (int c = 0; c < FDIM; ++c)
        o[c >> 3][c & 7] = (_Float16)fm[(long)c * VOXELS + v];
    h8* dst = (h8*)(ws + v * FDIM);
#pragma unroll
    for (int i = 0; i < 4; ++i) dst[i] = o[i];
}

// ---------------------------------------------------------------------------
// Pass 2: fused trilinear-sample + 2-layer MLP.
// Block = 256 threads (8 wave32), 256 points per block.
// ---------------------------------------------------------------------------
__global__ void __launch_bounds__(256)
nf_fused(const float* __restrict__ x, const float* __restrict__ fm,
         const _Float16* __restrict__ ws,
         const float* __restrict__ W1, const float* __restrict__ b1,
         const float* __restrict__ W2, const float* __restrict__ b2,
         float* __restrict__ out, int npts, int useWs) {
    __shared__ __align__(16) _Float16 exs[256 * EXSTRIDE];  // ex rows, f16, K-padded
    __shared__ __align__(16) _Float16 w1s[HID * EXSTRIDE];  // W1, f16, K-padded
    __shared__ float b1s[HID];
    __shared__ float w2s[HID];

    const int tid = threadIdx.x;
    const int blockStart = blockIdx.x * 256;

    // ---- stage W1 / b1 / W2 into LDS -------------------------------------
    for (int i = tid; i < HID * KPAD; i += 256) {
        int h = i >> 6, k = i & 63;
        w1s[h * EXSTRIDE + k] = (k < DIN) ? (_Float16)W1[h * DIN + k] : (_Float16)0.0f;
    }
    if (tid < HID) { b1s[tid] = b1[tid]; w2s[tid] = W2[tid]; }

    // ---- stage 1: trilinear sampling, one point per thread ----------------
    const int p = blockStart + tid;
    float feat[FDIM];
#pragma unroll
    for (int c = 0; c < FDIM; ++c) feat[c] = 0.0f;
    float px = 0.0f, py = 0.0f, pz = 0.0f;

    if (p < npts) {
        px = x[3 * p + 0]; py = x[3 * p + 1]; pz = x[3 * p + 2];
        const float ixf = (px * DINV + 1.0f) * 0.5f * (float)(GRIDN - 1);
        const float iyf = (py * DINV + 1.0f) * 0.5f * (float)(GRIDN - 1);
        const float izf = (pz * DINV + 1.0f) * 0.5f * (float)(GRIDN - 1);
        const float xfl = floorf(ixf), yfl = floorf(iyf), zfl = floorf(izf);
        const int x0 = (int)xfl, y0 = (int)yfl, z0 = (int)zfl;
        const float fx = ixf - xfl, fy = iyf - yfl, fz = izf - zfl;

#pragma unroll
        for (int k = 0; k < 8; ++k) {
            const int dz = (k >> 2) & 1, dy = (k >> 1) & 1, dx = k & 1;
            const float w = (dz ? fz : 1.0f - fz) * (dy ? fy : 1.0f - fy)
                          * (dx ? fx : 1.0f - fx);
            const int xi = min(max(x0 + dx, 0), GRIDN - 1);
            const int yi = min(max(y0 + dy, 0), GRIDN - 1);
            const int zi = min(max(z0 + dz, 0), GRIDN - 1);
            const long flat = ((long)zi * GRIDN + yi) * GRIDN + xi;
            if (useWs) {
                const h8* src = (const h8*)(ws + flat * FDIM);  // 64B contiguous
#pragma unroll
                for (int v = 0; v < 4; ++v) {
                    h8 hv = src[v];
#pragma unroll
                    for (int e = 0; e < 8; ++e)
                        feat[v * 8 + e] += w * (float)hv[e];
                }
            } else {  // fallback: strided gather from original channel-major fm
                const float* fp = fm + flat;
#pragma unroll
                for (int c = 0; c < FDIM; ++c)
                    feat[c] += w * fp[(long)c * VOXELS];
            }
        }
    }

    // write ex row = [x, y, z, feat0..31, 0 x 29] as f16
    {
        _Float16* row = &exs[tid * EXSTRIDE];
        row[0] = (_Float16)px; row[1] = (_Float16)py; row[2] = (_Float16)pz;
#pragma unroll
        for (int c = 0; c < FDIM; ++c) row[3 + c] = (_Float16)feat[c];
#pragma unroll
        for (int k = DIN; k < KPAD; ++k) row[k] = (_Float16)0.0f;
    }
    __syncthreads();

    // ---- stage 2: WMMA GEMM + fused bias/ReLU/layer-2 ---------------------
    const int lane = tid & 31;
    const int wave = tid >> 5;
    const int li = lane & 15;   // column (B/D) or row (A) index within tile
    const int hi = lane >> 4;   // half-wave selector
    const float b2v = b2[0];

#pragma unroll
    for (int sub = 0; sub < 2; ++sub) {
        const int mtile = wave * 2 + sub;          // 0..15 (16 rows each)
        const int arow  = mtile * 16 + li;

        // A-matrix (16x32 f16) per the CDNA5 layout:
        // lanes 0-15: K 0-7 (v0-3) & 16-23 (v4-7); lanes 16-31: K 8-15 & 24-31
        const h8* ar = (const h8*)&exs[arow * EXSTRIDE];  // 8 chunks of 8 halves
        v16h a0, a1;
        {
            h8 lo = ar[0 + hi], hp = ar[2 + hi];          // k-step 0
#pragma unroll
            for (int e = 0; e < 8; ++e) { a0[e] = lo[e]; a0[8 + e] = hp[e]; }
        }
        {
            h8 lo = ar[4 + hi], hp = ar[6 + hi];          // k-step 1
#pragma unroll
            for (int e = 0; e < 8; ++e) { a1[e] = lo[e]; a1[8 + e] = hp[e]; }
        }

        v8f pacc = {};  // per-lane partial sums of layer-2 dot product
#pragma unroll
        for (int t = 0; t < 8; ++t) {                     // 8 hidden-col tiles
            const int col = t * 16 + li;
            // B-matrix (32x16 f16): lane holds one column; lanes 0-15 K 0-7 & 16-23,
            // lanes 16-31 K 8-15 & 24-31 (same K order as A).
            const h8* br = (const h8*)&w1s[col * EXSTRIDE];
            v16h bv0, bv1;
            {
                h8 lo = br[0 + hi], hp = br[2 + hi];
#pragma unroll
                for (int e = 0; e < 8; ++e) { bv0[e] = lo[e]; bv0[8 + e] = hp[e]; }
            }
            {
                h8 lo = br[4 + hi], hp = br[6 + hi];
#pragma unroll
                for (int e = 0; e < 8; ++e) { bv1[e] = lo[e]; bv1[8 + e] = hp[e]; }
            }

            v8f acc = {};
            acc = __builtin_amdgcn_wmma_f32_16x16x32_f16(
                false, a0, false, bv0, (short)0, acc, false, false);
            acc = __builtin_amdgcn_wmma_f32_16x16x32_f16(
                false, a1, false, bv1, (short)0, acc, false, false);

            const float bias = b1s[col];
            const float w2v  = w2s[col];
#pragma unroll
            for (int j = 0; j < 8; ++j) {
                float hv = acc[j] + bias;     // D[j] = h[row mtile*16+j+8*hi][col]
                hv = fmaxf(hv, 0.0f);         // ReLU
                pacc[j] += hv * w2v;          // layer-2 partial
            }
        }

        // reduce over the 16 column-lanes of each half-wave
#pragma unroll
        for (int m = 1; m < 16; m <<= 1) {
#pragma unroll
            for (int j = 0; j < 8; ++j)
                pacc[j] += __shfl_xor(pacc[j], m, 32);
        }
        if (li == 0) {
            const int base = blockStart + mtile * 16 + hi * 8;
#pragma unroll
            for (int j = 0; j < 8; ++j) {
                const int r = base + j;
                if (r < npts) out[r] = pacc[j] + b2v;
            }
        }
    }
}

// ---------------------------------------------------------------------------
extern "C" void kernel_launch(void* const* d_in, const int* in_sizes, int n_in,
                              void* d_out, int out_size, void* d_ws, size_t ws_size,
                              hipStream_t stream) {
    const float* x  = (const float*)d_in[0];  // [N,3]
    const float* fm = (const float*)d_in[1];  // [32,129,129,129]
    const float* W1 = (const float*)d_in[2];  // [128,35]
    const float* b1 = (const float*)d_in[3];  // [128]
    const float* W2 = (const float*)d_in[4];  // [1,128]
    const float* b2 = (const float*)d_in[5];  // [1]
    float* out = (float*)d_out;

    const int npts = in_sizes[0] / 3;
    const size_t wsNeed = (size_t)VOXELS * FDIM * sizeof(_Float16);  // ~137 MB
    const int useWs = (ws_size >= wsNeed) ? 1 : 0;
    _Float16* ws = (_Float16*)d_ws;

    if (useWs) {
        const int tb = (int)((VOXELS + 255) / 256);
        fm_transpose_f16<<<tb, 256, 0, stream>>>(fm, ws);
    }
    const int nb = (npts + 255) / 256;
    nf_fused<<<nb, 256, 0, stream>>>(x, fm, ws, W1, b1, W2, b2, out, npts, useWs);
}
